// HopfieldMessageLayer_38706245272465
// MI455X (gfx1250) — matrix operand order repaired and checked by reading, not compiled, not run on hardware
//
#include <hip/hip_runtime.h>
#include <hip/hip_bf16.h>
#include <stddef.h>

// ---------------------------------------------------------------------------
// HopfieldMessageLayer for MI455X (gfx1250, wave32, WMMA bf16 16x16x32)
//   C=1024 tokens, B=8 batch, D=1024, H=16 heads, hd=64
// ---------------------------------------------------------------------------

#define CN 1024
#define BN 8
#define DN 1024
#define HN 16
#define HD 64
#define MTOT (CN * BN)   // 8192 rows
#define EPS 1e-5f
#define PPAD 33          // LDS row pitch (floats) to avoid bank conflicts

typedef __attribute__((ext_vector_type(16))) __bf16 v16bf;
typedef __attribute__((ext_vector_type(8)))  __bf16 v8bf;
typedef __attribute__((ext_vector_type(8)))  float  v8f;

union V16 { v16bf v; v8bf h[2]; };

#define WMMA_BF16(A, B_, C_) \
  __builtin_amdgcn_wmma_f32_16x16x32_bf16(false, (A), false, (B_), (short)0, (C_), false, false)

// ---------------------------------------------------------------------------
// fp32 -> bf16 convert (grid-stride)
// ---------------------------------------------------------------------------
__global__ __launch_bounds__(256) void cvt_f32_bf16(const float* __restrict__ x,
                                                    __bf16* __restrict__ y, int n) {
  for (int i = blockIdx.x * blockDim.x + threadIdx.x; i < n; i += gridDim.x * blockDim.x)
    y[i] = (__bf16)x[i];
}

// ---------------------------------------------------------------------------
// GEMM: Y[m,n] = sum_k X[m,k] * W[n,k]   (X: MTOT x DN bf16, W: DN x DN bf16)
// One wave computes a 32(M) x 64(N) tile, K-chunks of 32 (8 WMMA / chunk).
// MODE 0: store bf16 row-major                     (Q, K projections)
// MODE 1: store bf16 transposed to vT[b][h][d][c]  (V projection)
// MODE 2: store f32 + bias                         (output projection)
// ---------------------------------------------------------------------------
template <int MODE>
__global__ __launch_bounds__(32) void gemm_xwt(const __bf16* __restrict__ X,
                                               const __bf16* __restrict__ W,
                                               void* __restrict__ Y,
                                               const float* __restrict__ bias) {
  const int lane  = threadIdx.x;          // 0..31
  const int mBase = blockIdx.x * 32;
  const int nBase = blockIdx.y * 64;

  v8f acc0[4] = {};   // rows mBase .. mBase+15
  v8f acc1[4] = {};   // rows mBase+16 .. mBase+31

  // A: 16-bit A layout, contiguous K (matches B):
  //   lane<16 : K = {0..15}+kk ; lane>=16 : K = {16..31}+kk
  const __bf16* xrow0 = X + (size_t)(mBase + (lane & 15)) * DN;
  const __bf16* xrow1 = xrow0 + (size_t)16 * DN;
  const int kaOff = (lane < 16) ? 0 : 16;
  const int kbOff = (lane < 16) ? 0 : 16;

  for (int kk = 0; kk < DN; kk += 32) {
    V16 ua0, ua1;
    ua0.h[0] = *(const v8bf*)(xrow0 + kk + kaOff);
    ua0.h[1] = *(const v8bf*)(xrow0 + kk + kaOff + 8);
    ua1.h[0] = *(const v8bf*)(xrow1 + kk + kaOff);
    ua1.h[1] = *(const v8bf*)(xrow1 + kk + kaOff + 8);
    __builtin_prefetch(xrow0 + kk + 128, 0, 0);  // global_prefetch_b8

#pragma unroll
    for (int t = 0; t < 4; ++t) {
      // B column-major: lane holds column N = nBase + t*16 + (lane&15),
      // i.e. W row n, contiguous K (lane<16: kk..kk+15, lane>=16: kk+16..kk+31)
      const int n = nBase + t * 16 + (lane & 15);
      const v16bf bw = *(const v16bf*)(W + (size_t)n * DN + kk + kbOff);
      acc0[t] = WMMA_BF16(ua0.v, bw, acc0[t]);
      acc1[t] = WMMA_BF16(ua1.v, bw, acc1[t]);
    }
  }

  // C/D layout: lane<16 -> M=j, N=lane ; lane>=16 -> M=j+8, N=lane-16
  const int rOff = (lane < 16) ? 0 : 8;
  const int nl   = lane & 15;
#pragma unroll
  for (int half = 0; half < 2; ++half) {
    const v8f* acc = half ? acc1 : acc0;
#pragma unroll
    for (int t = 0; t < 4; ++t) {
#pragma unroll
      for (int j = 0; j < 8; ++j) {
        const int m = mBase + half * 16 + j + rOff;
        const int n = nBase + t * 16 + nl;
        if (MODE == 0) {
          ((__bf16*)Y)[(size_t)m * DN + n] = (__bf16)acc[t][j];
        } else if (MODE == 1) {
          const int c = m >> 3, b = m & 7;      // token m = c*B + b
          const int hh = n >> 6, d = n & 63;    // n = h*64 + d
          ((__bf16*)Y)[(((size_t)b * HN + hh) * HD + d) * CN + c] = (__bf16)acc[t][j];
        } else {
          ((float*)Y)[(size_t)m * DN + n] = acc[t][j] + bias[n];
        }
      }
    }
  }
}

// ---------------------------------------------------------------------------
// Flash attention: one wave = 16 queries of one (head h, batch b).
//   scores = beta * Q K^T, online softmax over keys, out = P V
// q,k : (MTOT, DN) bf16 ; vT : (B,H,HD,CN) bf16 ; ao : (MTOT, DN) bf16
//
// Softmax reduction strategy: scores go through LDS once (C-layout ->
// A-layout transpose). In A layout, row r's 32 scores live as 16 values in
// lane r and 16 in lane r+16, so the row max/sum is per-lane VALU plus a
// single shfl_xor(16); running stats are per-lane scalars. The exp'd values
// are already the A-matrix operand for the P@V WMMA.
// ---------------------------------------------------------------------------
__global__ __launch_bounds__(32) void attn_flash(const __bf16* __restrict__ q,
                                                 const __bf16* __restrict__ k,
                                                 const __bf16* __restrict__ vT,
                                                 const float* __restrict__ log_beta,
                                                 __bf16* __restrict__ ao) {
  const int qt   = blockIdx.x;   // query tile 0..63
  const int hh   = blockIdx.y;   // head
  const int bb   = blockIdx.z;   // batch
  const int lane = threadIdx.x;
  const float beta = __expf(log_beta[hh]);

  __shared__ float p32[16 * PPAD];

  // Preload Q tile as two A matrices (dims 0..31 and 32..63)
  const __bf16* qrow =
      q + ((size_t)(qt * 16 + (lane & 15)) * BN + bb) * DN + hh * HD;
  const int kaOff = (lane < 16) ? 0 : 16;
  V16 ua0, ua1;
  ua0.h[0] = *(const v8bf*)(qrow + kaOff);
  ua0.h[1] = *(const v8bf*)(qrow + kaOff + 8);
  ua1.h[0] = *(const v8bf*)(qrow + 32 + kaOff);
  ua1.h[1] = *(const v8bf*)(qrow + 32 + kaOff + 8);

  float mro = -1e30f;   // running max of row (lane&15)
  float lro = 0.0f;     // running sum of row (lane&15)
  v8f acc[4] = {};

  const int kbOff   = (lane < 16) ? 0 : 16;
  const int nl      = lane & 15;
  const int rOff    = (lane < 16) ? 0 : 8;   // C-layout row offset
  const int srcBase = rOff;                  // stat-owner lane offset

  for (int kk = 0; kk < CN; kk += 32) {
    // ---- scores: S0 = keys kk..kk+15, S1 = keys kk+16..kk+31 ---------------
    v8f s0 = {}, s1 = {};
#pragma unroll
    for (int half = 0; half < 2; ++half) {
      const int keyB = kk + half * 16;
      const __bf16* krow = k + ((size_t)(keyB + nl) * BN + bb) * DN + hh * HD;
      const v16bf bk0 = *(const v16bf*)(krow + kbOff);        // dims 0..31
      const v16bf bk1 = *(const v16bf*)(krow + 32 + kbOff);   // dims 32..63
      if (half == 0) {
        s0 = WMMA_BF16(ua0.v, bk0, s0);
        s0 = WMMA_BF16(ua1.v, bk1, s0);
      } else {
        s1 = WMMA_BF16(ua0.v, bk0, s1);
        s1 = WMMA_BF16(ua1.v, bk1, s1);
      }
    }

    // ---- C-layout -> LDS (scaled by beta) ----------------------------------
#pragma unroll
    for (int j = 0; j < 8; ++j) {
      p32[(j + rOff) * PPAD + nl]      = s0[j] * beta;
      p32[(j + rOff) * PPAD + 16 + nl] = s1[j] * beta;
    }
    __syncthreads();

    // ---- row-owner phase: read back in A layout ----------------------------
    // lane<16 : K = {0..15} ; lane>=16 : K = {16..31}
    float vb[16];
    {
      const int r  = lane & 15;
      const int kb = (lane < 16) ? 0 : 16;
#pragma unroll
      for (int e = 0; e < 8; ++e) vb[e]     = p32[r * PPAD + kb + e];
#pragma unroll
      for (int e = 0; e < 8; ++e) vb[8 + e] = p32[r * PPAD + kb + 8 + e];
    }
    __syncthreads();

    float pmax = vb[0];
#pragma unroll
    for (int i = 1; i < 16; ++i) pmax = fmaxf(pmax, vb[i]);
    pmax = fmaxf(pmax, __shfl_xor(pmax, 16));      // full row max
    const float mnew = fmaxf(mro, pmax);
    const float sc   = __expf(mro - mnew);         // rescale factor, row (lane&15)
    mro = mnew;

    V16 up;
    float psum = 0.0f;
#pragma unroll
    for (int i = 0; i < 16; ++i) {
      const float e = __expf(vb[i] - mnew);
      psum += e;
      up.v[i] = (__bf16)e;                         // already A-layout element i
    }
    psum += __shfl_xor(psum, 16);                  // full row sum
    lro = lro * sc + psum;

    // ---- rescale accumulators (element j <-> row j+rOff, stat at lane) -----
#pragma unroll
    for (int j = 0; j < 8; ++j) {
      const float scj = __shfl(sc, j + srcBase);
#pragma unroll
      for (int t = 0; t < 4; ++t) acc[t][j] *= scj;
    }

    // ---- P @ V : B columns are output dims, keys contiguous in vT ----------
#pragma unroll
    for (int t = 0; t < 4; ++t) {
      const __bf16* vrow =
          vT + (((size_t)bb * HN + hh) * HD + t * 16 + nl) * CN + kk + kbOff;
      const v16bf bv = *(const v16bf*)vrow;
      acc[t] = WMMA_BF16(up.v, bv, acc[t]);
    }
  }

  // ---- normalize and store -------------------------------------------------
#pragma unroll
  for (int j = 0; j < 8; ++j) {
    const float inv = 1.0f / __shfl(lro, j + srcBase);
    const int c = qt * 16 + j + rOff;
    __bf16* orow = ao + ((size_t)c * BN + bb) * DN + hh * HD;
#pragma unroll
    for (int t = 0; t < 4; ++t)
      orow[t * 16 + nl] = (__bf16)(acc[t][j] * inv);
  }
}

// ---------------------------------------------------------------------------
// LayerNorm over last dim (1024) : one 256-thread block per row
// ---------------------------------------------------------------------------
__global__ __launch_bounds__(256) void layernorm_rows(const float* __restrict__ x,
                                                      const float* __restrict__ gamma,
                                                      const float* __restrict__ beta,
                                                      float* __restrict__ y) {
  __shared__ float red[16];
  const int row = blockIdx.x;
  const int tid = threadIdx.x;
  const float* xr = x + (size_t)row * DN;

  float v[4], s = 0.0f, s2 = 0.0f;
#pragma unroll
  for (int i = 0; i < 4; ++i) {
    v[i] = xr[tid + i * 256];
    s += v[i];
    s2 += v[i] * v[i];
  }
#pragma unroll
  for (int m = 16; m >= 1; m >>= 1) {
    s  += __shfl_xor(s, m);
    s2 += __shfl_xor(s2, m);
  }
  const int wave = tid >> 5;
  if ((tid & 31) == 0) { red[wave] = s; red[8 + wave] = s2; }
  __syncthreads();
  if (tid < 32) {
    float a  = (tid < 8) ? red[tid] : 0.0f;
    float b2 = (tid < 8) ? red[8 + tid] : 0.0f;
#pragma unroll
    for (int m = 4; m >= 1; m >>= 1) {
      a  += __shfl_xor(a, m);
      b2 += __shfl_xor(b2, m);
    }
    if (tid == 0) { red[0] = a; red[1] = b2; }
  }
  __syncthreads();
  const float mu  = red[0] * (1.0f / DN);
  const float var = red[1] * (1.0f / DN) - mu * mu;
  const float inv = rsqrtf(var + EPS);
#pragma unroll
  for (int i = 0; i < 4; ++i) {
    const int col = tid + i * 256;
    y[(size_t)row * DN + col] = (v[i] - mu) * inv * gamma[col] + beta[col];
  }
}

// ---------------------------------------------------------------------------
// Orchestration
// ---------------------------------------------------------------------------
extern "C" void kernel_launch(void* const* d_in, const int* in_sizes, int n_in,
                              void* d_out, int out_size, void* d_ws, size_t ws_size,
                              hipStream_t stream) {
  const float* h        = (const float*)d_in[0];
  const float* W_q      = (const float*)d_in[1];
  const float* W_k      = (const float*)d_in[2];
  const float* W_v      = (const float*)d_in[3];
  const float* W_o      = (const float*)d_in[4];
  const float* b_o      = (const float*)d_in[5];
  const float* log_beta = (const float*)d_in[6];
  const float* ln_gamma = (const float*)d_in[7];
  const float* ln_beta  = (const float*)d_in[8];
  float* out = (float*)d_out;

  // Workspace carve-up (bf16 unless noted)
  char* ws = (char*)d_ws;
  __bf16* hb  = (__bf16*)ws;  ws += (size_t)MTOT * DN * 2;   // 16 MB
  __bf16* wqb = (__bf16*)ws;  ws += (size_t)DN * DN * 2;     //  2 MB
  __bf16* wkb = (__bf16*)ws;  ws += (size_t)DN * DN * 2;
  __bf16* wvb = (__bf16*)ws;  ws += (size_t)DN * DN * 2;
  __bf16* wob = (__bf16*)ws;  ws += (size_t)DN * DN * 2;
  __bf16* qb  = (__bf16*)ws;  ws += (size_t)MTOT * DN * 2;   // 16 MB
  __bf16* kb  = (__bf16*)ws;  ws += (size_t)MTOT * DN * 2;   // 16 MB
  __bf16* vTb = (__bf16*)ws;  ws += (size_t)MTOT * DN * 2;   // 16 MB (B,H,HD,C)
  __bf16* aob = (__bf16*)ws;  ws += (size_t)MTOT * DN * 2;   // 16 MB
  float*  pre = (float*)ws;   ws += (size_t)MTOT * DN * 4;   // 32 MB

  // 1) convert inputs to bf16
  cvt_f32_bf16<<<2048, 256, 0, stream>>>(h,   hb,  MTOT * DN);
  cvt_f32_bf16<<<1024, 256, 0, stream>>>(W_q, wqb, DN * DN);
  cvt_f32_bf16<<<1024, 256, 0, stream>>>(W_k, wkb, DN * DN);
  cvt_f32_bf16<<<1024, 256, 0, stream>>>(W_v, wvb, DN * DN);
  cvt_f32_bf16<<<1024, 256, 0, stream>>>(W_o, wob, DN * DN);

  // 2) Q/K/V projections (V stores transposed for attention B-matrix loads)
  const dim3 gGemm(MTOT / 32, DN / 64);
  gemm_xwt<0><<<gGemm, 32, 0, stream>>>(hb, wqb, (void*)qb,  nullptr);
  gemm_xwt<0><<<gGemm, 32, 0, stream>>>(hb, wkb, (void*)kb,  nullptr);
  gemm_xwt<1><<<gGemm, 32, 0, stream>>>(hb, wvb, (void*)vTb, nullptr);

  // 3) flash attention per (query-tile, head, batch)
  attn_flash<<<dim3(CN / 16, HN, BN), 32, 0, stream>>>(qb, kb, vTb, log_beta, aob);

  // 4) output projection + bias (f32)
  gemm_xwt<2><<<gGemm, 32, 0, stream>>>(aob, wob, (void*)pre, b_o);

  // 5) layernorm -> d_out
  layernorm_rows<<<MTOT, 256, 0, stream>>>(pre, ln_gamma, ln_beta, out);
}
